// LSTM_34600256536979
// MI455X (gfx1250) — matrix operand order repaired and checked by reading, not compiled, not run on hardware
//
#include <hip/hip_runtime.h>

// Problem constants (match reference)
#define Hd 1024
#define Td 100
#define Bd 32
#define FOURH 4096

typedef __attribute__((ext_vector_type(16))) __bf16 v16bf;
typedef __attribute__((ext_vector_type(8)))  __bf16 v8bf;
typedef __attribute__((ext_vector_type(8)))  float  v8f;

union AFrag { v16bf v; v8bf h[2]; };
struct Frags { AFrag a; v16bf b[4]; };

__device__ __forceinline__ float sigm(float x) { return 1.0f / (1.0f + __expf(-x)); }
__device__ __forceinline__ float tanh_f(float x) { return 2.0f / (1.0f + __expf(-2.0f * x)) - 1.0f; }

// Load one K-chunk (32 wide) of A fragment + 4 B fragments, in contiguous k order (k = 16*hi + s) for A and B.
__device__ __forceinline__ void load_frags(Frags& f, const __bf16* __restrict__ Arow,
                                           const __bf16* const* Wr, int kk, int hi)
{
    f.a.h[0] = *(const v8bf*)(Arow + kk + 16 * hi);
    f.a.h[1] = *(const v8bf*)(Arow + kk + 16 * hi + 8);
#pragma unroll
    for (int j = 0; j < 4; ++j) f.b[j] = *(const v16bf*)(Wr[j] + kk + 16 * hi);
}

__device__ __forceinline__ void do_wmma(v8f acc[4], const Frags& f)
{
#pragma unroll
    for (int j = 0; j < 4; ++j)
        acc[j] = __builtin_amdgcn_wmma_f32_16x16x32_bf16(
            false, f.a.v, false, f.b[j], (short)0, acc[j], false, false);
}

// Fully-unrolled K loop over 32 chunks (K = 1024). No back edge -> no
// loop-carried buffer registers -> scheduler software-pipelines the
// immediate-offset loads ahead of the WMMAs without rotation copies.
// Two alternating accumulator chains keep consecutive WMMA groups free of
// D-register RAW hazards.
__device__ __forceinline__ void k_loop(v8f acc[4], const __bf16* __restrict__ Arow,
                                       const __bf16* const* Wr, int hi)
{
    v8f a0[4] = {{}, {}, {}, {}};
    v8f a1[4] = {{}, {}, {}, {}};
#pragma unroll
    for (int ck = 0; ck < Hd / 32; ++ck) {
        Frags f;
        load_frags(f, Arow, Wr, ck * 32, hi);
        do_wmma((ck & 1) ? a1 : a0, f);
    }
#pragma unroll
    for (int j = 0; j < 4; ++j) acc[j] = a0[j] + a1[j];
}

// ---------------------------------------------------------------------------
// Big input-projection GEMM: out[M x 4096] = A[M x 1024](bf16) @ W[4096 x 1024]^T + bias
// One wave per (16-row, 64-col) job: 4 WMMA accumulators sharing one A fragment.
// grid: (M/16)*64 waves, 8 waves/block.
// ---------------------------------------------------------------------------
__global__ __launch_bounds__(256) void gemm_x_bias(
    const __bf16* __restrict__ A, const __bf16* __restrict__ W,
    const float* __restrict__ bias, float* __restrict__ out, int M)
{
    int tid  = blockIdx.x * 256 + threadIdx.x;
    int wid  = tid >> 5;
    int lane = threadIdx.x & 31;
    int mt = wid >> 6;      // 64 column-groups of 64 columns each
    int ng = wid & 63;
    int m0 = mt * 16, n0 = ng * 64;
    if (m0 >= M) return;
    int hi = lane >> 4, nl = lane & 15;

    const __bf16* Arow = A + (long long)(m0 + nl) * Hd;
    const __bf16* Wr[4];
#pragma unroll
    for (int j = 0; j < 4; ++j) Wr[j] = W + (long long)(n0 + j * 16 + nl) * Hd;

    v8f acc[4];
    k_loop(acc, Arow, Wr, hi);

#pragma unroll
    for (int j = 0; j < 4; ++j) {
        int n = n0 + j * 16 + nl;
        float bv = bias[n];
#pragma unroll
        for (int r = 0; r < 8; ++r) {
            int m = m0 + r + 8 * hi;
            out[(long long)m * FOURH + n] = acc[j][r] + bv;
        }
    }
}

// ---------------------------------------------------------------------------
// Fused recurrent step: gates = h_prev @ Whh^T + gates_x ; apply LSTM cell.
// Each wave owns one 16(batch) x 16(hidden) block: its i,f,g,o tiles share one
// A fragment (4 WMMA accumulators), then the full cell epilogue in registers.
// grid: 2 x 64 = 128 waves = 16 blocks of 256.
// ---------------------------------------------------------------------------
__global__ __launch_bounds__(256) void lstm_cell(
    const __bf16* __restrict__ hprev, long long hprevStride,
    const __bf16* __restrict__ Whh,
    const float*  __restrict__ gx,    long long gxStride,
    float*        __restrict__ c,
    __bf16*       __restrict__ hout,  long long houtStride,
    float*        __restrict__ fout,  long long foutStride)
{
    int tid  = blockIdx.x * 256 + threadIdx.x;
    int wid  = tid >> 5;
    int lane = threadIdx.x & 31;
    int mt = wid >> 6;
    int nt = wid & 63;
    int m0 = mt * 16, n0 = nt * 16;
    int hi = lane >> 4, nl = lane & 15;

    const __bf16* Arow = hprev + (long long)(m0 + nl) * hprevStride;
    const __bf16* Wr[4];
#pragma unroll
    for (int g = 0; g < 4; ++g) Wr[g] = Whh + (long long)(g * Hd + n0 + nl) * Hd;

    v8f acc[4];
    k_loop(acc, Arow, Wr, hi);

#pragma unroll
    for (int r = 0; r < 8; ++r) {
        int bb = m0 + r + 8 * hi;         // batch row
        int n  = n0 + nl;                 // hidden column
        const float* gxr = gx + (long long)bb * gxStride;
        float pi = acc[0][r] + gxr[n];
        float pf = acc[1][r] + gxr[Hd + n];
        float pg = acc[2][r] + gxr[2 * Hd + n];
        float po = acc[3][r] + gxr[3 * Hd + n];
        float iv = sigm(pi), fv = sigm(pf), gv = tanh_f(pg), ov = sigm(po);
        long long cidx = (long long)bb * Hd + n;
        float cn = fv * c[cidx] + iv * gv;
        c[cidx] = cn;
        float hv = ov * tanh_f(cn);
        hout[(long long)bb * houtStride + n] = (__bf16)hv;
        if (fout) fout[(long long)bb * foutStride + n] = hv;
    }
}

// ---------------------------------------------------------------------------
// Prep kernels
// ---------------------------------------------------------------------------
__global__ void cvt_f32_to_bf16(const float* __restrict__ s, __bf16* __restrict__ d, int n)
{
    int i = blockIdx.x * 256 + threadIdx.x;
    if (i < n) d[i] = (__bf16)s[i];
}

__global__ void combine_bias(const float* __restrict__ a0, const float* __restrict__ b0,
                             const float* __restrict__ a1, const float* __restrict__ b1,
                             float* __restrict__ out)
{
    int i = blockIdx.x * 256 + threadIdx.x;
    if (i < FOURH) { out[i] = a0[i] + b0[i]; out[FOURH + i] = a1[i] + b1[i]; }
}

__global__ void init_state(__bf16* __restrict__ zh, float* __restrict__ c0, float* __restrict__ c1)
{
    int i = blockIdx.x * 256 + threadIdx.x;
    if (i < Bd * Hd) { zh[i] = (__bf16)0.0f; c0[i] = 0.0f; c1[i] = 0.0f; }
}

// dec_in[:,0,:] = enc layer-1 h at t=T-1 (already bf16); dec_in[:,1:,:] = bf16(targets[:,1:,:])
__global__ void build_dec_in(const float* __restrict__ targets,
                             const __bf16* __restrict__ ench1,
                             __bf16* __restrict__ dec)
{
    int i = blockIdx.x * 256 + threadIdx.x;
    if (i >= Bd * Td * Hd) return;
    int n = i % Hd;
    int t = (i / Hd) % Td;
    int b = i / (Td * Hd);
    if (t == 0) dec[i] = ench1[(long long)b * Td * Hd + (long long)(Td - 1) * Hd + n];
    else        dec[i] = (__bf16)targets[i];
}

// ---------------------------------------------------------------------------
extern "C" void kernel_launch(void* const* d_in, const int* in_sizes, int n_in,
                              void* d_out, int out_size, void* d_ws, size_t ws_size,
                              hipStream_t stream)
{
    const float* x    = (const float*)d_in[0];
    const float* tgt  = (const float*)d_in[1];
    const float* Wih0 = (const float*)d_in[2];
    const float* Whh0 = (const float*)d_in[3];
    const float* bih0 = (const float*)d_in[4];
    const float* bhh0 = (const float*)d_in[5];
    const float* Wih1 = (const float*)d_in[6];
    const float* Whh1 = (const float*)d_in[7];
    const float* bih1 = (const float*)d_in[8];
    const float* bhh1 = (const float*)d_in[9];
    float* out = (float*)d_out;

    char* ws = (char*)d_ws;
    size_t off = 0;
    auto alloc = [&](size_t bytes) -> void* {
        void* p = ws + off;
        off += (bytes + 255) & ~(size_t)255;
        return p;
    };

    const size_t WN = (size_t)FOURH * Hd;      // 4M elements per weight matrix
    const size_t SN = (size_t)Bd * Td * Hd;    // 3.2M elements per sequence

    __bf16* wb_ih0 = (__bf16*)alloc(WN * 2);
    __bf16* wb_hh0 = (__bf16*)alloc(WN * 2);
    __bf16* wb_ih1 = (__bf16*)alloc(WN * 2);
    __bf16* wb_hh1 = (__bf16*)alloc(WN * 2);
    float*  bias2  = (float*) alloc(2 * FOURH * 4);
    __bf16* x0b    = (__bf16*)alloc(SN * 2);
    __bf16* decb   = (__bf16*)alloc(SN * 2);
    __bf16* enc_h0 = (__bf16*)alloc(SN * 2);
    __bf16* enc_h1 = (__bf16*)alloc(SN * 2);
    __bf16* dec_h0 = (__bf16*)alloc(SN * 2);
    __bf16* dec_h1 = (__bf16*)alloc(SN * 2);
    __bf16* zero_h = (__bf16*)alloc((size_t)Bd * Hd * 2);
    float*  c0     = (float*) alloc((size_t)Bd * Hd * 4);
    float*  c1     = (float*) alloc((size_t)Bd * Hd * 4);
    float*  gates  = (float*) alloc((size_t)Bd * Td * FOURH * 4);
    (void)ws_size; (void)in_sizes; (void)n_in; (void)out_size;

    // ---- prep ----
    cvt_f32_to_bf16<<<(int)((WN + 255) / 256), 256, 0, stream>>>(Wih0, wb_ih0, (int)WN);
    cvt_f32_to_bf16<<<(int)((WN + 255) / 256), 256, 0, stream>>>(Whh0, wb_hh0, (int)WN);
    cvt_f32_to_bf16<<<(int)((WN + 255) / 256), 256, 0, stream>>>(Wih1, wb_ih1, (int)WN);
    cvt_f32_to_bf16<<<(int)((WN + 255) / 256), 256, 0, stream>>>(Whh1, wb_hh1, (int)WN);
    cvt_f32_to_bf16<<<(int)((SN + 255) / 256), 256, 0, stream>>>(x, x0b, (int)SN);
    combine_bias<<<16, 256, 0, stream>>>(bih0, bhh0, bih1, bhh1, bias2);
    init_state<<<128, 256, 0, stream>>>(zero_h, c0, c1);

    const int M = Bd * Td;                 // 3200 rows
    const int gemmBlocks = (M / 16) * 64 / 8;  // 1600
    const long long sH  = (long long)Td * Hd;
    const long long sG  = (long long)Td * FOURH;

    // ---- encoder, layer 0 ----
    gemm_x_bias<<<gemmBlocks, 256, 0, stream>>>(x0b, wb_ih0, bias2, gates, M);
    for (int t = 0; t < Td; ++t) {
        const __bf16* hp = (t == 0) ? zero_h : enc_h0 + (size_t)(t - 1) * Hd;
        long long ps = (t == 0) ? (long long)Hd : sH;
        lstm_cell<<<16, 256, 0, stream>>>(hp, ps, wb_hh0,
                                          gates + (size_t)t * FOURH, sG, c0,
                                          enc_h0 + (size_t)t * Hd, sH, nullptr, 0);
    }
    // ---- encoder, layer 1 ----
    gemm_x_bias<<<gemmBlocks, 256, 0, stream>>>(enc_h0, wb_ih1, bias2 + FOURH, gates, M);
    for (int t = 0; t < Td; ++t) {
        const __bf16* hp = (t == 0) ? zero_h : enc_h1 + (size_t)(t - 1) * Hd;
        long long ps = (t == 0) ? (long long)Hd : sH;
        lstm_cell<<<16, 256, 0, stream>>>(hp, ps, wb_hh1,
                                          gates + (size_t)t * FOURH, sG, c1,
                                          enc_h1 + (size_t)t * Hd, sH, nullptr, 0);
    }
    // ---- decoder input ----
    build_dec_in<<<(int)((SN + 255) / 256), 256, 0, stream>>>(tgt, enc_h1, decb);

    // ---- decoder, layer 0 (h0/c0 carry over from encoder) ----
    gemm_x_bias<<<gemmBlocks, 256, 0, stream>>>(decb, wb_ih0, bias2, gates, M);
    for (int t = 0; t < Td; ++t) {
        const __bf16* hp = (t == 0) ? enc_h0 + (size_t)(Td - 1) * Hd : dec_h0 + (size_t)(t - 1) * Hd;
        lstm_cell<<<16, 256, 0, stream>>>(hp, sH, wb_hh0,
                                          gates + (size_t)t * FOURH, sG, c0,
                                          dec_h0 + (size_t)t * Hd, sH, nullptr, 0);
    }
    // ---- decoder, layer 1 -> writes f32 output ----
    gemm_x_bias<<<gemmBlocks, 256, 0, stream>>>(dec_h0, wb_ih1, bias2 + FOURH, gates, M);
    for (int t = 0; t < Td; ++t) {
        const __bf16* hp = (t == 0) ? enc_h1 + (size_t)(Td - 1) * Hd : dec_h1 + (size_t)(t - 1) * Hd;
        lstm_cell<<<16, 256, 0, stream>>>(hp, sH, wb_hh1,
                                          gates + (size_t)t * FOURH, sG, c1,
                                          dec_h1 + (size_t)t * Hd, sH,
                                          out + (size_t)t * Hd, sH);
    }
}
